// PyGT_TGCN_14181982011590
// MI455X (gfx1250) — matrix operand order repaired and checked by reading, not compiled, not run on hardware
//
#include <hip/hip_runtime.h>
#include <stdint.h>

typedef _Float16 half_t;
typedef __attribute__((ext_vector_type(16))) _Float16 v16h;
typedef __attribute__((ext_vector_type(8)))  float    v8f;

#define HD __device__ __forceinline__

HD v8f wmma32(v16h a, v16h b, v8f c) {
  // (neg_a, A, neg_b, B, c_mod, C, reuse_a, reuse_b)
  return __builtin_amdgcn_wmma_f32_16x16x32_f16(false, a, false, b, (short)0, c,
                                                false, false);
}

// ---- WMMA fragment layout helpers (wave32, 16x16x32 f16) ----
// A-matrix 16x32 (MxK): lane L -> m = L%16, h = L/16.
// element i: v=i/2, j=i%2; k = (v<4) ? h*8+v*2+j : 16+h*8+(v-4)*2+j
HD int a_k(int i, int h) {
  int v = i >> 1, j = i & 1;
  return (v < 4) ? (h * 8 + v * 2 + j) : (16 + h * 8 + (v - 4) * 2 + j);
}

HD v16h load_a_f32(const float* __restrict__ p, int ld, int m, int h, int k0) {
  v16h a;
#pragma unroll
  for (int i = 0; i < 16; ++i) a[i] = (_Float16)p[m * ld + k0 + a_k(i, h)];
  return a;
}

HD v16h load_a_f32_relu(const float* __restrict__ p, int ld, int m, int h, int k0) {
  v16h a;
#pragma unroll
  for (int i = 0; i < 16; ++i) {
    float x = p[m * ld + k0 + a_k(i, h)];
    a[i] = (_Float16)(x > 0.0f ? x : 0.0f);
  }
  return a;
}

HD v16h load_a_f16(const half_t* __restrict__ p, int ld, int m, int h, int k0) {
  v16h a;
#pragma unroll
  for (int i = 0; i < 16; ++i) a[i] = p[m * ld + k0 + a_k(i, h)];
  return a;
}

// B-matrix 32x16 (KxN): lane L -> n = n0 + L%16; element i -> k = k0 + a_k(i, L/16).
HD v16h load_b(const half_t* __restrict__ p, int ld, int k0, int n0, int lane) {
  v16h b;
  const half_t* col = p + (size_t)k0 * ld + n0 + (lane & 15);
#pragma unroll
  for (int i = 0; i < 16; ++i) b[i] = col[(size_t)a_k(i, lane >> 4) * ld];
  return b;
}

HD float sigmoidf_(float x) { return 1.0f / (1.0f + __expf(-x)); }

// ---------------- elementwise / scatter kernels ----------------

__global__ void k_cvt(const float* __restrict__ s, half_t* __restrict__ d, int n) {
  int i = blockIdx.x * blockDim.x + threadIdx.x;
  if (i < n) d[i] = (half_t)s[i];
}

__global__ void k_fill(float* __restrict__ p, float v, int n) {
  int i = blockIdx.x * blockDim.x + threadIdx.x;
  if (i < n) p[i] = v;
}

__global__ void k_degscatter(const int* __restrict__ dst, const float* __restrict__ w,
                             float* __restrict__ deg, int E_) {
  int e = blockIdx.x * blockDim.x + threadIdx.x;
  if (e < E_) atomicAdd(&deg[dst[e]], w[e]);
}

__global__ void k_dinv(float* __restrict__ deg, int n) {
  int i = blockIdx.x * blockDim.x + threadIdx.x;
  if (i < n) {
    float d = deg[i];
    deg[i] = (d > 0.0f) ? rsqrtf(d) : 0.0f;
  }
}

// agg_g[n][f] = bias_g[f] + dinv[n]^2 * xw_g[n][f]   (self-loop contribution)
__global__ void k_agg_init(const half_t* __restrict__ xw, const float* __restrict__ dinv,
                           const float* __restrict__ bz, const float* __restrict__ br,
                           const float* __restrict__ bh, float* __restrict__ agg, int N_) {
  size_t i = (size_t)blockIdx.x * blockDim.x + threadIdx.x;
  size_t NH = (size_t)N_ * 128;
  if (i < NH) {
    int node = (int)(i >> 7);
    int f = (int)(i & 127);
    float di = dinv[node];
    float c = di * di;
    agg[i]          = bz[f] + c * (float)xw[i];
    agg[NH + i]     = br[f] + c * (float)xw[NH + i];
    agg[2 * NH + i] = bh[f] + c * (float)xw[2 * NH + i];
  }
}

// one wave per edge; lane handles 4 features x 3 gates
__global__ void k_edges(const int* __restrict__ src, const int* __restrict__ dst,
                        const float* __restrict__ w, const float* __restrict__ dinv,
                        const half_t* __restrict__ xw, float* __restrict__ agg,
                        int E_, int N_) {
  int e = (blockIdx.x * blockDim.x + threadIdx.x) >> 5;
  int lane = threadIdx.x & 31;
  if (e >= E_) return;
  int s = src[e], d = dst[e];
  float nrm = dinv[s] * w[e] * dinv[d];
  size_t NH = (size_t)N_ * 128;
#pragma unroll
  for (int g = 0; g < 3; ++g) {
    const half_t* xg = xw + (size_t)g * NH + (size_t)s * 128;
    float* ag = agg + (size_t)g * NH + (size_t)d * 128;
#pragma unroll
    for (int i = 0; i < 4; ++i) {
      int f = lane + i * 32;
      atomicAdd(&ag[f], nrm * (float)xg[f]);
    }
  }
}

// ---------------- WMMA GEMM kernels ----------------

// xw_g = node_feat (Nx64) @ W_g (64x128), g=z,r,h ; output f16
__global__ void k_xw(const float* __restrict__ X, const half_t* __restrict__ Wg,
                     half_t* __restrict__ xw, int N_) {
  int wave = (blockIdx.x * blockDim.x + threadIdx.x) >> 5;
  int lane = threadIdx.x & 31;
  int tiles = N_ / 16;
  if (wave >= tiles) return;
  int row0 = wave * 16;
  int m = lane & 15, hh = lane >> 4;
  const float* Xr = X + (size_t)row0 * 64;
  v16h a0 = load_a_f32(Xr, 64, m, hh, 0);
  v16h a1 = load_a_f32(Xr, 64, m, hh, 32);
  size_t NH = (size_t)N_ * 128;
  for (int g = 0; g < 3; ++g) {
    const half_t* W = Wg + (size_t)g * 64 * 128;
    half_t* out = xw + (size_t)g * NH + (size_t)row0 * 128;
#pragma unroll
    for (int n = 0; n < 8; ++n) {
      v8f c = {};
      c = wmma32(a0, load_b(W, 128, 0, n * 16, lane), c);
      c = wmma32(a1, load_b(W, 128, 32, n * 16, lane), c);
      int col = n * 16 + (lane & 15);
#pragma unroll
      for (int r = 0; r < 8; ++r) out[(size_t)(r + 8 * hh) * 128 + col] = (half_t)c[r];
    }
  }
}

// Z = sigmoid([agg_z | Hs] @ Lzw + bz); R = sigmoid([agg_r | Hs] @ Lrw + br)
// stores Z (f32) and HsR = Hs*R (f16)
__global__ void k_zr(const float* __restrict__ aggz, const float* __restrict__ aggr,
                     const float* __restrict__ Hs, const half_t* __restrict__ Lzw,
                     const float* __restrict__ Lzb, const half_t* __restrict__ Lrw,
                     const float* __restrict__ Lrb, float* __restrict__ Z,
                     half_t* __restrict__ HsR, int N_) {
  int wave = (blockIdx.x * blockDim.x + threadIdx.x) >> 5;
  int lane = threadIdx.x & 31;
  int tiles = N_ / 16;
  if (wave >= tiles) return;
  int row0 = wave * 16;
  int m = lane & 15, hh = lane >> 4;
  const float* Az = aggz + (size_t)row0 * 128;
  const float* Ar = aggr + (size_t)row0 * 128;
  const float* Ah = Hs + (size_t)row0 * 128;
  v16h az[4], ar[4], ah[4];
#pragma unroll
  for (int c = 0; c < 4; ++c) {
    az[c] = load_a_f32(Az, 128, m, hh, c * 32);
    ar[c] = load_a_f32(Ar, 128, m, hh, c * 32);
    ah[c] = load_a_f32(Ah, 128, m, hh, c * 32);
  }
  for (int n = 0; n < 8; ++n) {
    v8f cz = {}, cr = {};
#pragma unroll
    for (int c = 0; c < 4; ++c) {
      cz = wmma32(az[c], load_b(Lzw, 128, c * 32, n * 16, lane), cz);
      cz = wmma32(ah[c], load_b(Lzw, 128, 128 + c * 32, n * 16, lane), cz);
      cr = wmma32(ar[c], load_b(Lrw, 128, c * 32, n * 16, lane), cr);
      cr = wmma32(ah[c], load_b(Lrw, 128, 128 + c * 32, n * 16, lane), cr);
    }
    int col = n * 16 + (lane & 15);
    float bzv = Lzb[col], brv = Lrb[col];
#pragma unroll
    for (int r = 0; r < 8; ++r) {
      int mm = r + 8 * hh;
      size_t idx = (size_t)(row0 + mm) * 128 + col;
      float zv = sigmoidf_(cz[r] + bzv);
      float rv = sigmoidf_(cr[r] + brv);
      Z[idx] = zv;
      HsR[idx] = (half_t)(Hs[idx] * rv);
    }
  }
}

// H_tilde = tanh([agg_h | Hs*R] @ Lhw + bh); h = Z*Hs + (1-Z)*H_tilde
__global__ void k_ht(const float* __restrict__ aggh, const float* __restrict__ Hs,
                     const half_t* __restrict__ HsR, const half_t* __restrict__ Lhw,
                     const float* __restrict__ Lhb, const float* __restrict__ Z,
                     float* __restrict__ h_out, int N_) {
  int wave = (blockIdx.x * blockDim.x + threadIdx.x) >> 5;
  int lane = threadIdx.x & 31;
  int tiles = N_ / 16;
  if (wave >= tiles) return;
  int row0 = wave * 16;
  int m = lane & 15, hh = lane >> 4;
  const float* A1 = aggh + (size_t)row0 * 128;
  const half_t* A2 = HsR + (size_t)row0 * 128;
  v16h a1[4], a2[4];
#pragma unroll
  for (int c = 0; c < 4; ++c) {
    a1[c] = load_a_f32(A1, 128, m, hh, c * 32);
    a2[c] = load_a_f16(A2, 128, m, hh, c * 32);
  }
  for (int n = 0; n < 8; ++n) {
    v8f acc = {};
#pragma unroll
    for (int c = 0; c < 4; ++c) {
      acc = wmma32(a1[c], load_b(Lhw, 128, c * 32, n * 16, lane), acc);
      acc = wmma32(a2[c], load_b(Lhw, 128, 128 + c * 32, n * 16, lane), acc);
    }
    int col = n * 16 + (lane & 15);
    float bv = Lhb[col];
#pragma unroll
    for (int r = 0; r < 8; ++r) {
      int mm = r + 8 * hh;
      size_t idx = (size_t)(row0 + mm) * 128 + col;
      float ht = tanhf(acc[r] + bv);
      float z = Z[idx];
      h_out[idx] = z * Hs[idx] + (1.0f - z) * ht;
    }
  }
}

// y = relu(h) @ Lout_w (128x64) + Lout_b
__global__ void k_yout(const float* __restrict__ h, const half_t* __restrict__ Low,
                       const float* __restrict__ Lob, float* __restrict__ y, int N_) {
  int wave = (blockIdx.x * blockDim.x + threadIdx.x) >> 5;
  int lane = threadIdx.x & 31;
  int tiles = N_ / 16;
  if (wave >= tiles) return;
  int row0 = wave * 16;
  int m = lane & 15, hh = lane >> 4;
  const float* A = h + (size_t)row0 * 128;
  v16h a[4];
#pragma unroll
  for (int c = 0; c < 4; ++c) a[c] = load_a_f32_relu(A, 128, m, hh, c * 32);
#pragma unroll
  for (int n = 0; n < 4; ++n) {
    v8f acc = {};
#pragma unroll
    for (int c = 0; c < 4; ++c)
      acc = wmma32(a[c], load_b(Low, 64, c * 32, n * 16, lane), acc);
    int col = n * 16 + (lane & 15);
    float bv = Lob[col];
#pragma unroll
    for (int r = 0; r < 8; ++r) {
      int mm = r + 8 * hh;
      y[(size_t)(row0 + mm) * 64 + col] = acc[r] + bv;
    }
  }
}

// ---------------- launcher ----------------

extern "C" void kernel_launch(void* const* d_in, const int* in_sizes, int n_in,
                              void* d_out, int out_size, void* d_ws, size_t ws_size,
                              hipStream_t stream) {
  const int* edge_index = (const int*)d_in[0];
  const float* node_feat = (const float*)d_in[1];
  const float* edge_weight = (const float*)d_in[2];
  const float* hidden = (const float*)d_in[3];
  const float* Wz = (const float*)d_in[4];  const float* bz = (const float*)d_in[5];
  const float* Wr = (const float*)d_in[6];  const float* br = (const float*)d_in[7];
  const float* Wh = (const float*)d_in[8];  const float* bh = (const float*)d_in[9];
  const float* Lz_w = (const float*)d_in[10]; const float* Lz_b = (const float*)d_in[11];
  const float* Lr_w = (const float*)d_in[12]; const float* Lr_b = (const float*)d_in[13];
  const float* Lh_w = (const float*)d_in[14]; const float* Lh_b = (const float*)d_in[15];
  const float* Lout_w = (const float*)d_in[16]; const float* Lout_b = (const float*)d_in[17];

  const int F_ = 64, H_ = 128;
  const int E_ = in_sizes[2];
  const int N_ = in_sizes[1] / F_;
  const size_t NH = (size_t)N_ * H_;

  const int* esrc = edge_index;
  const int* edst = edge_index + E_;

  // workspace carve-out
  uintptr_t base = (uintptr_t)d_ws;
  auto alloc = [&](size_t bytes) -> uintptr_t {
    uintptr_t p = base;
    base += (bytes + 255) & ~(size_t)255;
    return p;
  };
  float*  dinv = (float*)alloc((size_t)N_ * 4);
  half_t* xw   = (half_t*)alloc(3 * NH * 2);
  float*  agg  = (float*)alloc(3 * NH * 4);
  float*  Zg   = (float*)alloc(NH * 4);
  half_t* HsR  = (half_t*)alloc(NH * 2);
  half_t* Wf   = (half_t*)alloc((size_t)(3 * F_ * H_ + 3 * 2 * H_ * H_ + H_ * F_) * 2);
  half_t* Wzrh_h = Wf;                       // 3 * 64*128
  half_t* Lzw_h  = Wzrh_h + 3 * F_ * H_;
  half_t* Lrw_h  = Lzw_h + 2 * H_ * H_;
  half_t* Lhw_h  = Lrw_h + 2 * H_ * H_;
  half_t* Low_h  = Lhw_h + 2 * H_ * H_;

  float* y_out = (float*)d_out;              // N x 64
  float* h_out = y_out + (size_t)N_ * F_;    // N x 128

  const int TPB = 256;
  auto g1 = [&](size_t n) { return (int)((n + TPB - 1) / TPB); };

  // weight f32 -> f16 conversions
  k_cvt<<<g1(F_ * H_), TPB, 0, stream>>>(Wz, Wzrh_h, F_ * H_);
  k_cvt<<<g1(F_ * H_), TPB, 0, stream>>>(Wr, Wzrh_h + F_ * H_, F_ * H_);
  k_cvt<<<g1(F_ * H_), TPB, 0, stream>>>(Wh, Wzrh_h + 2 * F_ * H_, F_ * H_);
  k_cvt<<<g1(2 * H_ * H_), TPB, 0, stream>>>(Lz_w, Lzw_h, 2 * H_ * H_);
  k_cvt<<<g1(2 * H_ * H_), TPB, 0, stream>>>(Lr_w, Lrw_h, 2 * H_ * H_);
  k_cvt<<<g1(2 * H_ * H_), TPB, 0, stream>>>(Lh_w, Lhw_h, 2 * H_ * H_);
  k_cvt<<<g1(H_ * F_), TPB, 0, stream>>>(Lout_w, Low_h, H_ * F_);

  // GCN normalization: deg = 1 (self loop) + scatter(w, dst); dinv = rsqrt
  k_fill<<<g1(N_), TPB, 0, stream>>>(dinv, 1.0f, N_);
  k_degscatter<<<g1(E_), TPB, 0, stream>>>(edst, edge_weight, dinv, E_);
  k_dinv<<<g1(N_), TPB, 0, stream>>>(dinv, N_);

  const int tiles = N_ / 16;
  const int gemm_grid = (tiles + 3) / 4;   // 4 waves (128 threads) per block

  // xw_g = X @ W_g
  k_xw<<<gemm_grid, 128, 0, stream>>>(node_feat, Wzrh_h, xw, N_);

  // agg = bias + self-loop, then edge scatter
  k_agg_init<<<g1(NH), TPB, 0, stream>>>(xw, dinv, bz, br, bh, agg, N_);
  k_edges<<<(E_ + 7) / 8, 256, 0, stream>>>(esrc, edst, edge_weight, dinv, xw, agg,
                                            E_, N_);

  // gates
  k_zr<<<gemm_grid, 128, 0, stream>>>(agg, agg + NH, hidden, Lzw_h, Lz_b, Lrw_h,
                                      Lr_b, Zg, HsR, N_);
  k_ht<<<gemm_grid, 128, 0, stream>>>(agg + 2 * NH, hidden, HsR, Lhw_h, Lh_b, Zg,
                                      h_out, N_);
  k_yout<<<gemm_grid, 128, 0, stream>>>(h_out, Low_h, Lout_b, y_out, N_);

  (void)n_in; (void)out_size; (void)ws_size;
}
